// LSTM_27736898798264
// MI455X (gfx1250) — compile-verified, matrix operand order checked by reading, not run on hardware
//
#include <hip/hip_runtime.h>

// ---------------------------------------------------------------------------
// LSTM forward for MI455X (gfx1250, wave32, WMMA bf16 16x16x32).
// T=512, B=64, NIN=512, NH=1024, NOUT=512.
//
// Phase A (parallel prep): convert inputs to bf16; pack [W_x;W_h] and W_fo
//   into the exact WMMA B-fragment lane order (ISA 7.12.2, 16-bit B 32x16).
// Phase B (persistent scan): 16 WGs x 512 threads. Each wave owns one
//   (16 batch x 16 hidden) tile and computes all four gate preactivations
//   with v_wmma_f32_16x16x32_bf16 over K = 1536 using a branch-free 2-stage
//   ping-pong software pipeline. A-operand rows live in a fused global
//   buffer  A_t = [x_t | H_t]  (row stride 1536) so the K loop has a single
//   incremented base; x_{t+1} is copied into the next buffer cooperatively
//   at the top of each step (latency hidden under the GEMM). Cell update
//   fully in registers (C fp32 in WMMA D layout, exp-based sigmoid/tanh).
//   One device-wide atomic barrier per timestep. Output projection
//   H_{t+1} @ W_fo post-barrier by waves 0..7, same pipelining.
// ---------------------------------------------------------------------------

typedef __bf16 bf16;
typedef __attribute__((ext_vector_type(16))) __bf16 v16bf;
typedef __attribute__((ext_vector_type(8)))  float  v8f;

#define T_STEPS 512
#define BATCH   64
#define NIN     512
#define NH      1024
#define NOUT    512
#define KGATE   (NIN + NH)      // 1536
#define AROW    KGATE           // fused A-buffer row stride (elements)
#define KT_GATE (KGATE / 32)    // 48
#define KT_FO   (NH / 32)       // 32
#define NWG     16
#define THREADS 512             // 16 waves

// workspace layout (bytes)
#define OFF_XB   0u                    // T*B*NIN bf16 = 33554432 B
#define OFF_WG   33554432u             // 256 tiles * 48 kt * 512 bf16 = 12582912 B
#define OFF_WFO  46137344u             // 32 tiles * 32 kt * 512 bf16 = 1048576 B
#define OFF_ABUF 47185920u             // 2 * 64*1536 bf16 = 393216 B
#define OFF_CTR  47579136u

#define OUT_OFF_H ((size_t)T_STEPS * BATCH * NOUT)
#define OUT_OFF_C (OUT_OFF_H + (size_t)BATCH * NH)

__device__ __forceinline__ v8f zero8() {
  v8f z;
#pragma unroll
  for (int i = 0; i < 8; ++i) z[i] = 0.f;
  return z;
}

__device__ __forceinline__ float fast_sigmoid(float x) {
  return 1.f / (1.f + __expf(-x));            // v_exp_f32 + rcp
}
__device__ __forceinline__ float fast_tanh(float x) {
  return 1.f - 2.f / (__expf(2.f * x) + 1.f); // v_exp_f32 + rcp, no libdevice
}

// ISA 7.12.2: 16-bit fragment element e of lane (half = lane>>4) holds
// K = (e&7) + 8*half + 16*(e>=8)
__device__ __forceinline__ int frag_k(int lane, int e) {
  return (e & 7) + ((lane >> 4) << 3) + ((e & 8) << 1);
}

union Frag { v16bf v; uint4 u[2]; };

__device__ __forceinline__ void load_a_frag(Frag& f, const bf16* row, int half) {
  // lanes 0-15: K {0..7, 16..23}; lanes 16-31: K {8..15, 24..31}
  f.u[0] = *(const uint4*)(row + half * 8);
  f.u[1] = *(const uint4*)(row + 16 + half * 8);
}

__device__ __forceinline__ void load_b_frag(Frag& f, const bf16* tile, int lane) {
  const bf16* p = tile + lane * 16;     // pre-packed per-lane order
  f.u[0] = *(const uint4*)(p);
  f.u[1] = *(const uint4*)(p + 8);
}

// One pipeline stage of the gate GEMM: A fragment + 4 gate B fragments.
// Gate offsets are compile-time constants folded into load offsets.
__device__ __forceinline__ void load_stage(Frag& af, Frag (&bf)[4],
                                           const bf16* abase, const bf16* wbase,
                                           int kt, int lane, int half) {
  load_a_frag(af, abase + kt * 32, half);
  const bf16* wt = wbase + (size_t)kt * 512;
#pragma unroll
  for (int g = 0; g < 4; ++g)
    load_b_frag(bf[g], wt + (size_t)g * (KT_GATE * 512), lane);
}

__device__ __forceinline__ void wmma_gates(v8f (&acc)[4], const Frag& af,
                                           const Frag (&bf)[4]) {
#pragma unroll
  for (int g = 0; g < 4; ++g)
    acc[g] = __builtin_amdgcn_wmma_f32_16x16x32_bf16(
        false, af.v, false, bf[g].v, (short)0, acc[g], false, false);
}

// ------------------------------- prep kernels ------------------------------

__global__ void convert_x_kernel(const float* __restrict__ X, bf16* __restrict__ Xb, int n) {
  int stride = gridDim.x * blockDim.x;
  for (int i = blockIdx.x * blockDim.x + threadIdx.x; i < n; i += stride)
    Xb[i] = (bf16)X[i];
}

// Pack [W_x ; W_h] (K=1536) for all 4 gates into B-fragment tile order.
// Packed tile index p_tile = wg*16 + hidgroup*4 + gate; tile covers hidden
// units h = wg*64 + hidgroup*16 + (lane&15).
__global__ void pack_gate_weights_kernel(
    const float* __restrict__ Wxi, const float* __restrict__ Whi,
    const float* __restrict__ Wxf, const float* __restrict__ Whf,
    const float* __restrict__ Wxo, const float* __restrict__ Who,
    const float* __restrict__ Wxc, const float* __restrict__ Whc,
    bf16* __restrict__ Wg) {
  int tid = blockIdx.x * blockDim.x + threadIdx.x;
  if (tid >= 256 * KT_GATE * 32) return;
  int lane   = tid & 31;
  int kt     = (tid >> 5) % KT_GATE;
  int n_tile = tid / (KT_GATE * 32);
  int wgi    = n_tile >> 4;
  int local  = n_tile & 15;
  int hg     = local >> 2;
  int g      = local & 3;
  int h      = wgi * 64 + hg * 16 + (lane & 15);
  const float* Wx = (g == 0) ? Wxi : (g == 1) ? Wxf : (g == 2) ? Wxo : Wxc;
  const float* Wh = (g == 0) ? Whi : (g == 1) ? Whf : (g == 2) ? Who : Whc;
  bf16* dst = Wg + ((size_t)n_tile * KT_GATE + kt) * 512 + lane * 16;
#pragma unroll
  for (int e = 0; e < 16; ++e) {
    int k = kt * 32 + frag_k(lane, e);
    float val = (k < NIN) ? Wx[(size_t)k * NH + h] : Wh[(size_t)(k - NIN) * NH + h];
    dst[e] = (bf16)val;
  }
}

__global__ void pack_fo_weights_kernel(const float* __restrict__ Wfo, bf16* __restrict__ dst) {
  int tid = blockIdx.x * blockDim.x + threadIdx.x;
  if (tid >= 32 * KT_FO * 32) return;
  int lane   = tid & 31;
  int kt     = (tid >> 5) & 31;
  int n_tile = tid >> 10;
  int col    = n_tile * 16 + (lane & 15);
  bf16* d = dst + ((size_t)n_tile * KT_FO + kt) * 512 + lane * 16;
#pragma unroll
  for (int e = 0; e < 16; ++e) {
    int k = kt * 32 + frag_k(lane, e);
    d[e] = (bf16)Wfo[(size_t)k * NOUT + col];
  }
}

// Zero both fused A buffers, place x_0 into buffer 0's x-region, reset ctr.
__global__ void init_state_kernel(const bf16* __restrict__ Xb,
                                  bf16* __restrict__ Abuf,
                                  unsigned* __restrict__ ctr) {
  int stride = gridDim.x * blockDim.x;
  int n = 2 * BATCH * AROW;
  for (int i = blockIdx.x * blockDim.x + threadIdx.x; i < n; i += stride) {
    int buf = i / (BATCH * AROW);
    int rem = i % (BATCH * AROW);
    int r = rem / AROW;
    int c = rem % AROW;
    bf16 v = (bf16)0.f;
    if (buf == 0 && c < NIN) v = Xb[(size_t)r * NIN + c];
    Abuf[i] = v;
  }
  if (blockIdx.x == 0 && threadIdx.x == 0) *ctr = 0u;
}

// ---------------------------- persistent scan ------------------------------

__global__ void __launch_bounds__(THREADS)
lstm_scan_kernel(const bf16* __restrict__ Xb,
                 const bf16* __restrict__ Wg,
                 const bf16* __restrict__ Wfo,
                 bf16* __restrict__ Abuf,
                 const float* __restrict__ bI, const float* __restrict__ bF,
                 const float* __restrict__ bO, const float* __restrict__ bC,
                 const float* __restrict__ bFO,
                 float* __restrict__ out,
                 unsigned* __restrict__ ctr) {
  const int wg   = blockIdx.x;        // 0..15 : owns hidden units [wg*64, wg*64+64)
  const int tid  = threadIdx.x;
  const int wave = tid >> 5;          // 0..15
  const int lane = tid & 31;
  const int half = lane >> 4;
  const int ln   = lane & 15;

  const int m_tile = wave & 3;        // batch tile (16 rows)
  const int hg     = wave >> 2;       // hidden group (16 units)
  const int arow_m = m_tile * 16 + ln;        // A-fragment row (batch index)
  const int h      = wg * 64 + hg * 16 + ln;  // this lane's hidden unit (D col)
  const int ptile0 = wg * 16 + hg * 4;        // packed B tile base (+gate)

  const float biasI = bI[h], biasF = bF[h], biasO = bO[h], biasC = bC[h];
  const bf16* wbase = Wg + (size_t)ptile0 * (KT_GATE * 512);

  // x-copy assignment (4 rows per WG, 8 bytes per thread)
  const int xr = wg * 4 + (tid >> 7);
  const int xc = (tid & 127) * 4;

  v8f cstate = zero8();               // cell state, fp32, WMMA D layout
  unsigned phase = 0;

  for (int t = 0; t < T_STEPS; ++t) {
    const bf16* At    = Abuf + (size_t)(t & 1) * (BATCH * AROW);
    bf16*       Anext = Abuf + (size_t)((t + 1) & 1) * (BATCH * AROW);

    // ---- cooperative copy of x_{t+1}: issued first, hidden under GEMM ----
    if (t + 1 < T_STEPS) {
      const bf16* xsrc = Xb + (size_t)(t + 1) * (BATCH * NIN);
      *(uint2*)(Anext + (size_t)xr * AROW + xc) =
          *(const uint2*)(xsrc + (size_t)xr * NIN + xc);
    }

    // ---- gate preactivations: A_t @ Wg  (M=16, N=4x16, K=1536) ----
    v8f acc[4];
#pragma unroll
    for (int g = 0; g < 4; ++g) acc[g] = zero8();

    const bf16* abase = At + (size_t)arow_m * AROW;
    Frag afA, afB;
    Frag bfA[4], bfB[4];
    load_stage(afA, bfA, abase, wbase, 0, lane, half);      // kt = 0 in flight
#pragma unroll 1
    for (int kt = 0; kt < KT_GATE - 2; kt += 2) {
      load_stage(afB, bfB, abase, wbase, kt + 1, lane, half);
      __builtin_prefetch(wbase + (size_t)(kt + 2) * 512, 0, 1);
      wmma_gates(acc, afA, bfA);
      load_stage(afA, bfA, abase, wbase, kt + 2, lane, half);
      wmma_gates(acc, afB, bfB);
    }
    load_stage(afB, bfB, abase, wbase, KT_GATE - 1, lane, half);  // kt = 47
    wmma_gates(acc, afA, bfA);                                    // kt = 46
    wmma_gates(acc, afB, bfB);                                    // kt = 47

    // ---- LSTM cell update, all in registers (WMMA D layout) ----
    v8f hval;
#pragma unroll
    for (int v = 0; v < 8; ++v) {
      float ig = fast_sigmoid(acc[0][v] + biasI);
      float fg = fast_sigmoid(acc[1][v] + biasF);
      float og = fast_sigmoid(acc[2][v] + biasO);
      float ch = fast_tanh(acc[3][v] + biasC);
      float c  = fg * cstate[v] + ig * ch;
      cstate[v] = c;
      hval[v]  = og * fast_tanh(c);
    }
#pragma unroll
    for (int v = 0; v < 8; ++v) {
      int r = m_tile * 16 + v + half * 8;   // D layout: M = v + 8*half
      Anext[(size_t)r * AROW + NIN + h] = (bf16)hval[v];
    }
    if (t == T_STEPS - 1) {
#pragma unroll
      for (int v = 0; v < 8; ++v) {
        int r = m_tile * 16 + v + half * 8;
        out[OUT_OFF_H + (size_t)r * NH + h] = hval[v];
        out[OUT_OFF_C + (size_t)r * NH + h] = cstate[v];
      }
    }

    // ---- device-wide barrier: all 16 WGs finished writing Anext ----
    ++phase;
    __threadfence();
    __syncthreads();
    if (tid == 0) {
      __hip_atomic_fetch_add(ctr, 1u, __ATOMIC_RELEASE, __HIP_MEMORY_SCOPE_AGENT);
      while (__hip_atomic_load(ctr, __ATOMIC_ACQUIRE, __HIP_MEMORY_SCOPE_AGENT) <
             phase * (unsigned)NWG)
        __builtin_amdgcn_s_sleep(1);
    }
    __syncthreads();
    __threadfence();

    // ---- output projection: out[t] = H_{t+1} @ Wfo + b_fo ----
    if (wave < 8) {
      const int mt    = wave & 3;
      const int ftile = wg * 2 + (wave >> 2);       // global 16-col tile, 0..31
      const int orow  = mt * 16 + ln;
      const bf16* hrow    = Anext + (size_t)orow * AROW + NIN;
      const bf16* wfobase = Wfo + (size_t)ftile * (KT_FO * 512);
      v8f oacc = zero8();
      Frag oaA, obA, oaB, obB;
      load_a_frag(oaA, hrow, half);
      load_b_frag(obA, wfobase, lane);
#pragma unroll 1
      for (int kt = 0; kt < KT_FO - 2; kt += 2) {
        load_a_frag(oaB, hrow + (kt + 1) * 32, half);
        load_b_frag(obB, wfobase + (size_t)(kt + 1) * 512, lane);
        oacc = __builtin_amdgcn_wmma_f32_16x16x32_bf16(
            false, oaA.v, false, obA.v, (short)0, oacc, false, false);
        load_a_frag(oaA, hrow + (kt + 2) * 32, half);
        load_b_frag(obA, wfobase + (size_t)(kt + 2) * 512, lane);
        oacc = __builtin_amdgcn_wmma_f32_16x16x32_bf16(
            false, oaB.v, false, obB.v, (short)0, oacc, false, false);
      }
      load_a_frag(oaB, hrow + (KT_FO - 1) * 32, half);
      load_b_frag(obB, wfobase + (size_t)(KT_FO - 1) * 512, lane);
      oacc = __builtin_amdgcn_wmma_f32_16x16x32_bf16(
          false, oaA.v, false, obA.v, (short)0, oacc, false, false);
      oacc = __builtin_amdgcn_wmma_f32_16x16x32_bf16(
          false, oaB.v, false, obB.v, (short)0, oacc, false, false);
      const int col = ftile * 16 + ln;
      const float bias = bFO[col];
      float* obase = out + (size_t)t * (BATCH * NOUT);
#pragma unroll
      for (int v = 0; v < 8; ++v) {
        int r = mt * 16 + v + half * 8;
        obase[(size_t)r * NOUT + col] = oacc[v] + bias;
      }
    }
  }
}

// ------------------------------- launch ------------------------------------

extern "C" void kernel_launch(void* const* d_in, const int* in_sizes, int n_in,
                              void* d_out, int out_size, void* d_ws, size_t ws_size,
                              hipStream_t stream) {
  (void)in_sizes; (void)n_in; (void)out_size; (void)ws_size;
  const float* X   = (const float*)d_in[0];
  const float* Wxi = (const float*)d_in[1];
  const float* Whi = (const float*)d_in[2];
  const float* bI  = (const float*)d_in[3];
  const float* Wxf = (const float*)d_in[4];
  const float* Whf = (const float*)d_in[5];
  const float* bF  = (const float*)d_in[6];
  const float* Wxo = (const float*)d_in[7];
  const float* Who = (const float*)d_in[8];
  const float* bO  = (const float*)d_in[9];
  const float* Wxc = (const float*)d_in[10];
  const float* Whc = (const float*)d_in[11];
  const float* bC  = (const float*)d_in[12];
  const float* Wfo = (const float*)d_in[13];
  const float* bFO = (const float*)d_in[14];

  char* ws = (char*)d_ws;
  bf16*     Xb   = (bf16*)(ws + OFF_XB);
  bf16*     Wg   = (bf16*)(ws + OFF_WG);
  bf16*     Wfop = (bf16*)(ws + OFF_WFO);
  bf16*     Abuf = (bf16*)(ws + OFF_ABUF);
  unsigned* ctr  = (unsigned*)(ws + OFF_CTR);

  const int nx = T_STEPS * BATCH * NIN;
  convert_x_kernel<<<4096, 256, 0, stream>>>(X, Xb, nx);

  const int ngw = 256 * KT_GATE * 32;
  pack_gate_weights_kernel<<<(ngw + 255) / 256, 256, 0, stream>>>(
      Wxi, Whi, Wxf, Whf, Wxo, Who, Wxc, Whc, Wg);

  const int nfw = 32 * KT_FO * 32;
  pack_fo_weights_kernel<<<(nfw + 255) / 256, 256, 0, stream>>>(Wfo, Wfop);

  init_state_kernel<<<192, 256, 0, stream>>>(Xb, Abuf, ctr);

  lstm_scan_kernel<<<NWG, THREADS, 0, stream>>>(
      Xb, Wg, Wfop, Abuf, bI, bF, bO, bC, bFO, (float*)d_out, ctr);
}
